// PACE_nodagseq_58205396795637
// MI455X (gfx1250) — compile-verified, matrix operand order checked by reading, not run on hardware
//
#include <hip/hip_runtime.h>
#include <hip/hip_bf16.h>

typedef __attribute__((ext_vector_type(16))) _Float16 v16h;
typedef __attribute__((ext_vector_type(8)))  _Float16 h8;
typedef __attribute__((ext_vector_type(8)))  float    v8f;

#define B_      1024
#define MAXN    32
#define NINP    256
#define NHID    512
#define NHEAD   8
#define DH      64
#define NL      6
#define NZ      64
#define EPS_    1e-5f
#define SCALE_  0.125f      // 1/sqrt(64)

// ---------------------------------------------------------------------------
// WMMA fragment helpers (16-bit A/B layout per CDNA5 ISA §7.12.2):
// lane<16: M/N = lane,    K = k0+{0..7} and k0+{16..23}
// lane>=16: M/N = lane-16, K = k0+8+{0..7} and k0+8+{16..23}  (k0 = 32-blk base)
// Operands are stored so each fragment needs two contiguous 16B loads.
// ---------------------------------------------------------------------------
__device__ __forceinline__ v16h ldfrag16(const _Float16* p) {
  union { v16h v; struct { h8 lo; h8 hi; } s; } u;
  u.s.lo = *reinterpret_cast<const h8*>(p);
  u.s.hi = *reinterpret_cast<const h8*>(p + 16);
  return u.v;
}

// D(16x16,f32) += A(16xK from row-major A, stride sA) * B(Kx16 from Bt = B^T
// row-major, stride sB).  K multiple of 32.
__device__ __forceinline__ v8f tile_mm(const _Float16* A, int sA, int m0,
                                       const _Float16* Bt, int sB, int n0,
                                       int K, int lane) {
  const int hh  = (lane >> 4) & 1;
  const int l15 = lane & 15;
  const _Float16* ap = A  + (size_t)(m0 + l15) * sA + hh * 8;
  const _Float16* bp = Bt + (size_t)(n0 + l15) * sB + hh * 8;
  v8f acc = {0.f, 0.f, 0.f, 0.f, 0.f, 0.f, 0.f, 0.f};
  for (int kb = 0; kb < K; kb += 32) {
    v16h a = ldfrag16(ap + kb);
    v16h b = ldfrag16(bp + kb);
    acc = __builtin_amdgcn_wmma_f32_16x16x32_f16(
        /*neg_a=*/false, a, /*neg_b=*/false, b,
        /*c_mod=*/(short)0, acc, /*reuse_a=*/false, /*reuse_b=*/false);
  }
  return acc;
}

// y[32x512] = A[32x512] @ W[512x512] + bias, W given transposed f16 (Wt[n][k]).
// 64 output tiles split over 8 waves.
//   out16 != null : store f16 (optionally transposed for V, ReLU for FFN1)
//   acc32 != null : accumulate into f32 buffer (fused residual: x += y)
// Tiles are disjoint per wave, so the += has no write conflicts.
__device__ __forceinline__ void mm32x512(const _Float16* A, const _Float16* Wt,
                                         const float* bias,
                                         _Float16* out16, float* acc32,
                                         bool relu, bool transpose,
                                         int lane, int wave) {
  const int hh  = (lane >> 4) & 1;
  const int l15 = lane & 15;
  for (int t = wave; t < 64; t += 8) {
    const int m0 = (t & 1) * 16;
    const int n0 = (t >> 1) * 16;
    v8f acc = tile_mm(A, NHID, m0, Wt, NHID, n0, NHID, lane);
    const int col = n0 + l15;
    const float bv = bias[col];
#pragma unroll
    for (int r = 0; r < 8; ++r) {
      const int row = m0 + hh * 8 + r;            // C/D layout: VGPR r -> M=r(+8)
      float v = acc[r] + bv;
      if (relu) v = fmaxf(v, 0.f);
      if (acc32) acc32[row * NHID + col] += v;    // fused residual add
      if (out16) {
        if (transpose) out16[col * MAXN + row] = (_Float16)v;
        else           out16[row * NHID + col] = (_Float16)v;
      }
    }
  }
}

// In-place LayerNorm of xf32 (already holds x + y); refresh f16 WMMA copy.
__device__ __forceinline__ void ln_inplace(float* xf32, _Float16* xf16,
                                           const float* g, const float* bb,
                                           float* red, float* red2, float* mv,
                                           int tid) {
  const int row = tid >> 3;
  const int sub = tid & 7;
  const int c0  = sub * 64;
  float s = 0.f, s2 = 0.f;
  for (int c = c0; c < c0 + 64; ++c) {
    float v = xf32[row * NHID + c];
    s += v; s2 += v * v;
  }
  red [row * 8 + sub] = s;
  red2[row * 8 + sub] = s2;
  __syncthreads();
  if (sub == 0) {
    float ts = 0.f, ts2 = 0.f;
    for (int i = 0; i < 8; ++i) { ts += red[row * 8 + i]; ts2 += red2[row * 8 + i]; }
    float mean = ts * (1.f / NHID);
    float var  = ts2 * (1.f / NHID) - mean * mean;
    mv[row * 2 + 0] = mean;
    mv[row * 2 + 1] = rsqrtf(var + EPS_);
  }
  __syncthreads();
  const float mean = mv[row * 2 + 0];
  const float rstd = mv[row * 2 + 1];
  for (int c = c0; c < c0 + 64; ++c) {
    float v = (xf32[row * NHID + c] - mean) * rstd * g[c] + bb[c];
    xf32[row * NHID + c] = v;
    xf16[row * NHID + c] = (_Float16)v;
  }
  __syncthreads();
}

// ---------------------------------------------------------------------------
// Prep: fp32 [NL][512][512] (k-major) -> f16 transposed [NL][512][512] (n-major)
// ---------------------------------------------------------------------------
extern "C" __global__ void __launch_bounds__(256)
pace_convert_w(const float* __restrict__ src, _Float16* __restrict__ dst) {
  size_t t = (size_t)blockIdx.x * 256 + threadIdx.x;    // NL*512*512 total
  int k = (int)(t & 511);
  int n = (int)((t >> 9) & 511);
  int l = (int)(t >> 18);
  dst[((size_t)l * 512 + n) * 512 + k] =
      (_Float16)src[((size_t)l * 512 + k) * 512 + n];
}

// ---------------------------------------------------------------------------
// Fused encoder: one workgroup (8 wave32s) per graph; all 6 layers in LDS.
// ---------------------------------------------------------------------------
#define SMEM_BYTES 248320

extern "C" __global__ void __launch_bounds__(256, 1)
pace_encoder(const int* __restrict__ ntypes, const int* __restrict__ adj,
             const float* __restrict__ Wpos, const float* __restrict__ bpos,
             const float* __restrict__ Wnode, const float* __restrict__ bnode,
             const _Float16* __restrict__ Wt,   // [6 mats][NL][512][512] f16 (n,k)
             const float* __restrict__ bq, const float* __restrict__ bk,
             const float* __restrict__ bv, const float* __restrict__ bo,
             const float* __restrict__ b1, const float* __restrict__ b2,
             const float* __restrict__ ln1g, const float* __restrict__ ln1b,
             const float* __restrict__ ln2g, const float* __restrict__ ln2b,
             const float* __restrict__ Wfc1, const float* __restrict__ bfc1,
             const float* __restrict__ Wfc2, const float* __restrict__ bfc2,
             float* __restrict__ out) {
  extern __shared__ char smem[];
  float*    xf32  = (float*)   (smem);             // 65536  x master / residual acc
  _Float16* xf16  = (_Float16*)(smem + 65536);     // 32768  x (WMMA A)
  _Float16* qbuf  = (_Float16*)(smem + 98304);     // 32768  Q, later attn-out
  _Float16* kbuf  = (_Float16*)(smem + 131072);    // 32768  K, later FFN hidden
  _Float16* vtb   = (_Float16*)(smem + 163840);    // 32768  V^T [512][32]
  float*    sbuf  = (float*)   (smem + 196608);    // 32768  scores [8][32][32]
  _Float16* pbuf  = (_Float16*)(smem + 229376);    // 16384  probs  [8][32][32]
  float*    red   = (float*)   (smem + 245760);    // 1024
  float*    red2  = (float*)   (smem + 246784);    // 1024
  float*    mv    = (float*)   (smem + 247808);    // 256
  unsigned* amask = (unsigned*)(smem + 248064);    // 128    allowed-key bitmask
  unsigned* radj  = (unsigned*)(smem + 248192);    // 128    reachability rows

  const int b    = blockIdx.x;
  const int tid  = threadIdx.x;
  const int lane = tid & 31;
  const int wave = tid >> 5;

  // ---- embedding: x = concat(relu(Wnode[type]+bnode), Wpos+bpos) ----
  for (int idx = tid; idx < MAXN * NHID; idx += 256) {
    int n = idx >> 9, c = idx & 511;
    float v;
    if (c < NINP) {
      int tn = ntypes[b * MAXN + n];
      v = fmaxf(Wnode[tn * NINP + c] + bnode[c], 0.f);
    } else {
      int cc = c - NINP;
      v = Wpos[n * NINP + cc] + bpos[cc];
    }
    xf32[idx] = v;
    xf16[idx] = (_Float16)v;
  }

  // ---- DAG transitive closure as 32-bit masks (5 doubling rounds) ----
  if (tid < MAXN) {
    unsigned r = 0;
    for (int j = tid + 1; j < MAXN; ++j)
      if (adj[((size_t)b * MAXN + tid) * MAXN + j] != 0) r |= (1u << j);
    radj[tid] = r;
  }
  __syncthreads();
  for (int it = 0; it < 5; ++it) {
    unsigned nr = 0;
    if (tid < MAXN) {
      nr = radj[tid];
      unsigned m = nr;
      while (m) {
        int j = __ffs((int)m) - 1;
        m &= m - 1;
        nr |= radj[j];
      }
    }
    __syncthreads();
    if (tid < MAXN) radj[tid] = nr;
    __syncthreads();
  }
  if (tid < MAXN) {                       // allowed[q][k] = reach[k][q] || q==k
    unsigned a = 1u << tid;
    for (int k = 0; k < MAXN; ++k)
      if ((radj[k] >> tid) & 1u) a |= (1u << k);
    amask[tid] = a;
  }
  __syncthreads();

  const int hh  = (lane >> 4) & 1;
  const int l15 = lane & 15;

  for (int l = 0; l < NL; ++l) {
    const size_t MS = (size_t)NL * 512 * 512;
    const _Float16* WtQ = Wt + 0 * MS + (size_t)l * 512 * 512;
    const _Float16* WtK = Wt + 1 * MS + (size_t)l * 512 * 512;
    const _Float16* WtV = Wt + 2 * MS + (size_t)l * 512 * 512;
    const _Float16* WtO = Wt + 3 * MS + (size_t)l * 512 * 512;
    const _Float16* Wt1 = Wt + 4 * MS + (size_t)l * 512 * 512;
    const _Float16* Wt2 = Wt + 5 * MS + (size_t)l * 512 * 512;

    // Q / K / V projections (V stored transposed for the AV matmul)
    mm32x512(xf16, WtQ, bq + l * NHID, qbuf, nullptr, false, false, lane, wave);
    mm32x512(xf16, WtK, bk + l * NHID, kbuf, nullptr, false, false, lane, wave);
    mm32x512(xf16, WtV, bv + l * NHID, vtb,  nullptr, false, true,  lane, wave);
    __syncthreads();

    // scores s = Q K^T * SCALE + dag_bias   (per head: 32x32, K=64)
    for (int t = wave; t < 32; t += 8) {
      int h  = t >> 2;
      int mi = (t >> 1) & 1;
      int ni = t & 1;
      v8f acc = tile_mm(qbuf + h * DH, NHID, mi * 16,
                        kbuf + h * DH, NHID, ni * 16, DH, lane);
      int kcol = ni * 16 + l15;
#pragma unroll
      for (int r = 0; r < 8; ++r) {
        int q = mi * 16 + hh * 8 + r;
        float sv = acc[r] * SCALE_;
        if (!((amask[q] >> kcol) & 1u)) sv = -1e30f;
        sbuf[(h * MAXN + q) * MAXN + kcol] = sv;
      }
    }
    __syncthreads();

    // softmax over keys: one (head,query) row per thread
    {
      int h = tid >> 5, q = tid & 31;
      float* srow = sbuf + (h * MAXN + q) * MAXN;
      float mx = srow[0];
      for (int k = 1; k < MAXN; ++k) mx = fmaxf(mx, srow[k]);
      float ssum = 0.f;
      for (int k = 0; k < MAXN; ++k) {
        float e = __expf(srow[k] - mx);
        srow[k] = e;
        ssum += e;
      }
      float inv = 1.f / ssum;
      _Float16* prow = pbuf + (h * MAXN + q) * MAXN;
      for (int k = 0; k < MAXN; ++k) prow[k] = (_Float16)(srow[k] * inv);
    }
    __syncthreads();

    // o = P @ V  (per head: 32x64, K=32) -> qbuf (Q no longer needed)
    for (int t = wave; t < 64; t += 8) {
      int h  = t >> 3;
      int mi = (t >> 2) & 1;
      int nd = t & 3;
      v8f acc = tile_mm(pbuf + h * MAXN * MAXN, MAXN, mi * 16,
                        vtb, MAXN, h * DH + nd * 16, MAXN, lane);
      int col = h * DH + nd * 16 + l15;
#pragma unroll
      for (int r = 0; r < 8; ++r) {
        int row = mi * 16 + hh * 8 + r;
        qbuf[row * NHID + col] = (_Float16)acc[r];
      }
    }
    __syncthreads();

    // output projection with fused residual (xf32 += o@Wo + bo), then LN1
    mm32x512(qbuf, WtO, bo + l * NHID, nullptr, xf32, false, false, lane, wave);
    __syncthreads();
    ln_inplace(xf32, xf16, ln1g + l * NHID, ln1b + l * NHID, red, red2, mv, tid);

    // FFN: h = relu(x W1 + b1) -> kbuf ; xf32 += h W2 + b2 ; LN2
    mm32x512(xf16, Wt1, b1 + l * NHID, kbuf, nullptr, true, false, lane, wave);
    __syncthreads();
    mm32x512(kbuf, Wt2, b2 + l * NHID, nullptr, xf32, false, false, lane, wave);
    __syncthreads();
    ln_inplace(xf32, xf16, ln2g + l * NHID, ln2b + l * NHID, red, red2, mv, tid);
  }

  // ---- output heads: z1 = mem @ Wfc1 + bfc1 ; z2 = mem @ Wfc2 + bfc2 ----
  {
    int o  = tid >> 1;         // 0..127: first 64 -> fc1, next 64 -> fc2
    int h2 = tid & 1;
    int c  = o & 63;
    const float* W = (o < NZ) ? Wfc1 : Wfc2;
    float s = 0.f;
    int i0 = h2 * 8192;
    for (int i = i0; i < i0 + 8192; ++i)
      s += xf32[i] * W[(size_t)i * NZ + c];
    red[tid] = s;
    __syncthreads();
    if (h2 == 0) {
      float tot = red[2 * o] + red[2 * o + 1];
      if (o < NZ) out[(size_t)b * NZ + c] = tot + bfc1[c];
      else        out[(size_t)B_ * NZ + (size_t)b * NZ + c] = tot + bfc2[c];
    }
  }
}

// ---------------------------------------------------------------------------
extern "C" void kernel_launch(void* const* d_in, const int* in_sizes, int n_in,
                              void* d_out, int out_size, void* d_ws,
                              size_t ws_size, hipStream_t stream) {
  (void)in_sizes; (void)n_in; (void)out_size; (void)ws_size;
  const int*   ntypes = (const int*)  d_in[0];
  const int*   adj    = (const int*)  d_in[1];
  const float* Wpos   = (const float*)d_in[2];
  const float* bpos   = (const float*)d_in[3];
  const float* Wnode  = (const float*)d_in[4];
  const float* bnode  = (const float*)d_in[5];
  // d_in[6..11] = enc_{Wq,Wk,Wv,Wo,W1,W2}  [NL,512,512] fp32
  const float* bq   = (const float*)d_in[12];
  const float* bk   = (const float*)d_in[13];
  const float* bv   = (const float*)d_in[14];
  const float* bo   = (const float*)d_in[15];
  const float* b1   = (const float*)d_in[16];
  const float* b2   = (const float*)d_in[17];
  const float* ln1g = (const float*)d_in[18];
  const float* ln1b = (const float*)d_in[19];
  const float* ln2g = (const float*)d_in[20];
  const float* ln2b = (const float*)d_in[21];
  const float* Wfc1 = (const float*)d_in[22];
  const float* bfc1 = (const float*)d_in[23];
  const float* Wfc2 = (const float*)d_in[24];
  const float* bfc2 = (const float*)d_in[25];

  _Float16* wt = (_Float16*)d_ws;     // [6][NL][512][512] f16, transposed
  const size_t per_mat = (size_t)NL * 512 * 512;
  const int cvt_blocks = (int)(per_mat / 256);
  for (int mat = 0; mat < 6; ++mat) {
    pace_convert_w<<<cvt_blocks, 256, 0, stream>>>(
        (const float*)d_in[6 + mat], wt + (size_t)mat * per_mat);
  }

  pace_encoder<<<B_, 256, SMEM_BYTES, stream>>>(
      ntypes, adj, Wpos, bpos, Wnode, bnode, wt,
      bq, bk, bv, bo, b1, b2, ln1g, ln1b, ln2g, ln2b,
      Wfc1, bfc1, Wfc2, bfc2, (float*)d_out);
}
